// CustomLinear_48146583388712
// MI455X (gfx1250) — matrix operand order repaired and checked by reading, not compiled, not run on hardware
//
#include <hip/hip_runtime.h>

// ---------------------------------------------------------------------------
// CDNA5 (gfx1250) WMMA types
// ---------------------------------------------------------------------------
typedef __attribute__((ext_vector_type(16))) __bf16 v16bf;
typedef __attribute__((ext_vector_type(8)))  float  v8f;

// A-matrix fragment (16x32 bf16, row-major LDS tile, row stride 32):
// lane = half*16 + row; elements i=0..7 -> K = half*8 + i (VGPR0-3),
// i=8..15 -> K = 16 + half*8 + (i-8) (VGPR4-7). Two contiguous 16B runs.
static __device__ __forceinline__ v16bf frag_a(const __bf16* lds, int row, int half) {
  union { v16bf v; uint4 q[2]; } u;
  u.q[0] = *(const uint4*)(lds + row * 32 + half * 8);
  u.q[1] = *(const uint4*)(lds + row * 32 + 16 + half * 8);
  return u.v;
}
// B-matrix fragment (32x16 bf16 = K x N), stored as B^T row-major [n][32 k]:
// same K order as frag_a: i=0..7 -> K = half*8 + i, i=8..15 -> K = 16 + half*8 + (i-8).
static __device__ __forceinline__ v16bf frag_b(const __bf16* lds, int n, int half) {
  union { v16bf v; uint4 q[2]; } u;
  u.q[0] = *(const uint4*)(lds + n * 32 + half * 8);
  u.q[1] = *(const uint4*)(lds + n * 32 + 16 + half * 8);
  return u.v;
}

static __device__ __forceinline__ v8f wmma_bf16(v16bf a, v16bf b, v8f c) {
  return __builtin_amdgcn_wmma_f32_16x16x32_bf16(false, a, false, b, (short)0, c, false, false);
}

// CDNA5 async global->LDS copy (16B per lane), tracked by ASYNCcnt.
// the LDS address comes from the VDST VGPR; generic LDS pointers carry the LDS byte
// offset in their low 32 bits.
static __device__ __forceinline__ void async_load_b128(void* lds_ptr, const void* gptr) {
  unsigned lds_off = (unsigned)(unsigned long long)lds_ptr;
  asm volatile("global_load_async_to_lds_b128 %0, %1, off"
               :: "v"(lds_off), "v"(gptr) : "memory");
}
static __device__ __forceinline__ void wait_asynccnt0() {
  asm volatile("s_wait_asynccnt 0" ::: "memory");
}

// ---------------------------------------------------------------------------
// C[M,N] = A[M,K] * B[N,K]^T (+bias). bf16 operands, f32 accumulate.
// A is a virtual column-concat: col c < splitA -> A0, else A1 (+offA1).
// Block: 256 threads = 8 waves, 128x64 tile; wave grid 4(M) x 2(N);
// each wave: 2x2 WMMA subtiles with fragment reuse. grid = (N/64, M/128).
// Double-buffered LDS, async global->LDS copies overlap the WMMAs.
// Outputs: C (f32, nullable), Cb (bf16, nullable).
// ---------------------------------------------------------------------------
__global__ void __launch_bounds__(256)
gemm_abt_wmma(const __bf16* __restrict__ A0, int ldA0,
              const __bf16* __restrict__ A1, int ldA1, int splitA, int offA1,
              const __bf16* __restrict__ B, int ldB,
              float* __restrict__ C, __bf16* __restrict__ Cb, int ldC,
              int K, const float* __restrict__ bias)
{
  __shared__ __align__(16) __bf16 ldsA[2][128 * 32];
  __shared__ __align__(16) __bf16 ldsB[2][64 * 32];
  const int tid  = threadIdx.x;
  const int lane = tid & 31;
  const int wv   = tid >> 5;
  const int wm   = wv >> 1;            // 0..3
  const int wn   = wv & 1;             // 0..1
  const int m0 = blockIdx.y * 128;
  const int n0 = blockIdx.x * 64;
  const int ar  = tid >> 1;            // 0..127 : A row within tile
  const int akh = (tid & 1) * 16;      // 0/16   : A k-half (16 bf16 = 32B)
  const int br  = tid >> 2;            // 0..63  : B row within tile
  const int bkq = (tid & 3) * 8;       // 0..24  : B k-quarter (8 bf16 = 16B)
  v8f acc[2][2] = {};

  auto issue_tile = [&](int buf, int k0) {
    const int c = k0 + akh;
    const __bf16* pa = (c < splitA)
        ? (A0 + (size_t)(m0 + ar) * ldA0 + c)
        : (A1 + (size_t)(m0 + ar) * ldA1 + (c - splitA) + offA1);
    async_load_b128(&ldsA[buf][ar * 32 + akh],     pa);
    async_load_b128(&ldsA[buf][ar * 32 + akh + 8], pa + 8);
    const int cb = k0 + bkq;
    async_load_b128(&ldsB[buf][br * 32 + bkq], B + (size_t)(n0 + br) * ldB + cb);
  };

  issue_tile(0, 0);
  int buf = 0;
  for (int k0 = 0; k0 < K; k0 += 32) {
    wait_asynccnt0();        // own async copies into ldsX[buf] complete
    __syncthreads();         // ... and everyone else's too
    if (k0 + 32 < K) issue_tile(buf ^ 1, k0 + 32);   // overlap with WMMAs
    v16bf af0 = frag_a(ldsA[buf], wm * 32 + (lane & 15),      lane >> 4);
    v16bf af1 = frag_a(ldsA[buf], wm * 32 + 16 + (lane & 15), lane >> 4);
    v16bf bf0 = frag_b(ldsB[buf], wn * 32 + (lane & 15),      lane >> 4);
    v16bf bf1 = frag_b(ldsB[buf], wn * 32 + 16 + (lane & 15), lane >> 4);
    acc[0][0] = wmma_bf16(af0, bf0, acc[0][0]);
    acc[0][1] = wmma_bf16(af0, bf1, acc[0][1]);
    acc[1][0] = wmma_bf16(af1, bf0, acc[1][0]);
    acc[1][1] = wmma_bf16(af1, bf1, acc[1][1]);
    __syncthreads();         // all reads of ldsX[buf] done before its reuse
    buf ^= 1;
  }

  // D layout: VGPR r -> M = r + 8*(lane>>4), N = lane&15
#pragma unroll
  for (int mi = 0; mi < 2; ++mi)
#pragma unroll
    for (int ni = 0; ni < 2; ++ni) {
      const int rowb = m0 + wm * 32 + mi * 16 + (lane >> 4) * 8;
      const int col  = n0 + wn * 32 + ni * 16 + (lane & 15);
      const float badd = bias ? bias[col] : 0.0f;
#pragma unroll
      for (int r = 0; r < 8; ++r) {
        float v = acc[mi][ni][r] + badd;
        if (C)  C[(size_t)(rowb + r) * ldC + col] = v;
        if (Cb) Cb[(size_t)(rowb + r) * ldC + col] = (__bf16)v;
      }
    }
}

// ---------------------------------------------------------------------------
// G[256,256] = Cx^T Cx, Cx = [A0 (128 cols bf16) | A1 block (128 cols bf16)].
// Block: 8 waves splitting the row (K) range; partials reduced through LDS.
// grid (16,16); rows must be a multiple of 256.
// ---------------------------------------------------------------------------
__global__ void __launch_bounds__(256)
gram_wmma(const __bf16* __restrict__ A0, int ldA0,
          const __bf16* __restrict__ A1, int ldA1, int offA1,
          float* __restrict__ G, int rows)
{
  __shared__ __align__(16) __bf16 ldsA[8][16 * 32];
  __shared__ __align__(16) __bf16 ldsB[8][16 * 32];
  __shared__ float red[8 * 256];
  const int tid  = threadIdx.x;
  const int lane = tid & 31;
  const int wv   = tid >> 5;
  const int i0 = blockIdx.y * 16;
  const int j0 = blockIdx.x * 16;
  const __bf16* pi = (i0 < 128) ? (A0 + i0) : (A1 + (i0 - 128) + offA1);
  const __bf16* pj = (j0 < 128) ? (A0 + j0) : (A1 + (j0 - 128) + offA1);
  const int ldi = (i0 < 128) ? ldA0 : ldA1;
  const int ldj = (j0 < 128) ? ldA0 : ldA1;
  v8f acc = {};
  for (int r0 = wv * 32; r0 < rows; r0 += 256) {
    const size_t r = (size_t)(r0 + lane);
    {
      union { uint4 q[2]; __bf16 h[16]; } u;
      const uint4* pa = (const uint4*)(pi + r * ldi);
      u.q[0] = pa[0]; u.q[1] = pa[1];
#pragma unroll
      for (int t = 0; t < 16; ++t) ldsA[wv][t * 32 + lane] = u.h[t];   // transpose
    }
    {
      union { uint4 q[2]; __bf16 h[16]; } u;
      const uint4* pb = (const uint4*)(pj + r * ldj);
      u.q[0] = pb[0]; u.q[1] = pb[1];
#pragma unroll
      for (int t = 0; t < 16; ++t) ldsB[wv][t * 32 + lane] = u.h[t];
    }
    v16bf af = frag_a(ldsA[wv], lane & 15, lane >> 4);
    v16bf bf = frag_b(ldsB[wv], lane & 15, lane >> 4);
    acc = wmma_bf16(af, bf, acc);
  }
#pragma unroll
  for (int rr = 0; rr < 8; ++rr) red[wv * 256 + rr * 32 + lane] = acc[rr];
  __syncthreads();
  {
    float s = 0.0f;
#pragma unroll
    for (int w2 = 0; w2 < 8; ++w2) s += red[w2 * 256 + tid];
    const int rr = tid >> 5, ln = tid & 31;
    const int row = i0 + rr + 8 * (ln >> 4);
    const int col = j0 + (ln & 15);
    G[(size_t)row * 256 + col] = s;
  }
}

// ---------------------------------------------------------------------------
// Cholesky (upper) of G[256,256] -> R (f32 scratch) + triangular inverse,
// both also emitted as bf16 for downstream WMMA. Single workgroup.
// ---------------------------------------------------------------------------
__global__ void __launch_bounds__(256)
chol_inv(const float* __restrict__ G, float* __restrict__ R, float* __restrict__ Rinv,
         __bf16* __restrict__ Rb, __bf16* __restrict__ Rib)
{
  const int tid = threadIdx.x;
  __shared__ float rowj[256];
  __shared__ float sdiag;
  for (int idx = tid; idx < 256 * 256; idx += 256) {
    int i = idx >> 8, j = idx & 255;
    R[idx] = (i <= j) ? G[idx] : 0.0f;
  }
  __syncthreads();
  for (int j = 0; j < 256; ++j) {
    if (tid == 0) sdiag = sqrtf(fmaxf(R[j * 256 + j], 1e-30f));
    __syncthreads();
    const float invd = 1.0f / sdiag;
    {
      int c = j + tid;
      if (c < 256) {
        float v = (c == j) ? sdiag : R[j * 256 + c] * invd;
        R[j * 256 + c] = v;
        rowj[c] = v;
      }
    }
    __syncthreads();
    for (int a = j + 1; a < 256; ++a) {
      int c = a + tid;
      if (c < 256) R[a * 256 + c] -= rowj[a] * rowj[c];
    }
    __syncthreads();
  }
  // Back-substitution: thread `tid` owns column j=tid of Rinv.
  const int j = tid;
  for (int i = j + 1; i < 256; ++i) Rinv[i * 256 + j] = 0.0f;
  Rinv[j * 256 + j] = 1.0f / R[j * 256 + j];
  for (int i = j - 1; i >= 0; --i) {
    float s = 0.0f;
    for (int k = i + 1; k <= j; ++k) s += R[i * 256 + k] * Rinv[k * 256 + j];
    Rinv[i * 256 + j] = -s / R[i * 256 + i];
  }
  __syncthreads();
  for (int idx = tid; idx < 256 * 256; idx += 256) {
    Rb[idx]  = (__bf16)R[idx];
    Rib[idx] = (__bf16)Rinv[idx];
  }
}

// ---------------------------------------------------------------------------
// One-sided Jacobi SVD of Mc[256,256]; outputs (sorted, shrunk, scaled, bf16):
//   WsTb[j,k] = U[k, perm_j] * s_j   (U = W/S, so scale = s_j/S_j), j < 128
//   VsTb[j,k] = V[k, perm_j] * s_j
// Wt/Vt are f32 column-major work arrays. Single workgroup.
// ---------------------------------------------------------------------------
__global__ void __launch_bounds__(256)
jacobi_svd(const float* __restrict__ Mc, float* __restrict__ Wt, float* __restrict__ Vt,
           __bf16* __restrict__ WsTb, __bf16* __restrict__ VsTb)
{
  const int tid = threadIdx.x;
  for (int idx = tid; idx < 256 * 256; idx += 256) {
    int cI = idx >> 8, r = idx & 255;
    Wt[idx] = Mc[r * 256 + cI];
    Vt[idx] = (r == cI) ? 1.0f : 0.0f;
  }
  __syncthreads();
  for (int sweep = 0; sweep < 8; ++sweep) {
    for (int round = 0; round < 255; ++round) {
      if (tid < 128) {
        int a = (tid == 0) ? 0 : (1 + (tid - 1 + round) % 255);
        int b = 1 + (254 - tid + round) % 255;
        int P = a < b ? a : b, Q = a < b ? b : a;
        float* wp = Wt + (size_t)P * 256;
        float* wq = Wt + (size_t)Q * 256;
        float app = 0.f, aqq = 0.f, apq = 0.f;
        for (int r = 0; r < 256; ++r) {
          float xp = wp[r], xq = wq[r];
          app += xp * xp; aqq += xq * xq; apq += xp * xq;
        }
        if (fabsf(apq) > 1e-20f) {
          float tau = (aqq - app) / (2.0f * apq);
          float t = ((tau >= 0.0f) ? 1.0f : -1.0f) / (fabsf(tau) + sqrtf(1.0f + tau * tau));
          float c = 1.0f / sqrtf(1.0f + t * t);
          float s = t * c;
          float* vp = Vt + (size_t)P * 256;
          float* vq = Vt + (size_t)Q * 256;
          for (int r = 0; r < 256; ++r) {
            float xp = wp[r], xq = wq[r];
            wp[r] = c * xp - s * xq;
            wq[r] = s * xp + c * xq;
            float yp = vp[r], yq = vq[r];
            vp[r] = c * yp - s * yq;
            vq[r] = s * yp + c * yq;
          }
        }
      }
      __syncthreads();
    }
  }
  __shared__ float sS[256];
  __shared__ int   sIdx[256];
  {
    const float* wr = Wt + (size_t)tid * 256;
    float acc = 0.f;
    for (int r = 0; r < 256; ++r) acc += wr[r] * wr[r];
    sS[tid] = sqrtf(acc);
    sIdx[tid] = tid;
  }
  __syncthreads();
  for (int phase = 0; phase < 256; ++phase) {
    if (tid < 255 && ((tid & 1) == (phase & 1))) {
      if (sS[tid] < sS[tid + 1]) {
        float tf = sS[tid]; sS[tid] = sS[tid + 1]; sS[tid + 1] = tf;
        int   ti = sIdx[tid]; sIdx[tid] = sIdx[tid + 1]; sIdx[tid + 1] = ti;
      }
    }
    __syncthreads();
  }
  __shared__ float scW[128], scV[128];
  if (tid < 128) {
    float delta = sS[128];                       // sigma_{ell+1}
    float sv = sS[tid];
    float sh = sqrtf(fmaxf(sv - delta, 0.0f));   // symmetric split
    scV[tid] = sh;
    scW[tid] = (sv > 1e-20f) ? (sh / sv) : 0.0f;
  }
  __syncthreads();
  for (int jj = 0; jj < 128; ++jj) {
    int src = sIdx[jj];
    WsTb[(size_t)jj * 256 + tid] = (__bf16)(Wt[(size_t)src * 256 + tid] * scW[jj]);
    VsTb[(size_t)jj * 256 + tid] = (__bf16)(Vt[(size_t)src * 256 + tid] * scV[jj]);
  }
}

// ---------------------------------------------------------------------------
// Elementwise f32 -> bf16 conversion.
// ---------------------------------------------------------------------------
__global__ void cvt_bf16(const float* __restrict__ src, __bf16* __restrict__ dst, int count)
{
  int idx = blockIdx.x * blockDim.x + threadIdx.x;
  if (idx < count) dst[idx] = (__bf16)src[idx];
}

// FD fill phase: dst[r, 0:128] = (bf16) src[r, 0:128]
__global__ void copy_block128_bf16(const float* __restrict__ src, int ldSrc,
                                   __bf16* __restrict__ dst, int total)
{
  int idx = blockIdx.x * blockDim.x + threadIdx.x;
  if (idx >= total) return;
  int r = idx >> 7, c = idx & 127;
  dst[idx] = (__bf16)src[(size_t)r * ldSrc + c];
}

// ---------------------------------------------------------------------------
extern "C" void kernel_launch(void* const* d_in, const int* in_sizes, int n_in,
                              void* d_out, int out_size, void* d_ws, size_t ws_size,
                              hipStream_t stream) {
  const float* x    = (const float*)d_in[0];   // [m, n]
  const float* w    = (const float*)d_in[1];   // [p, n]
  const float* bias = (const float*)d_in[2];   // [p]
  float* out = (float*)d_out;                  // [m, p]

  const int p = in_sizes[2];                   // 5120
  const int n = in_sizes[1] / p;               // 1280
  const int m = in_sizes[0] / n;               // 8192
  const int ell = n / 10;                      // 128
  const int nb = n / ell;                      // 10

  char* wsp = (char*)d_ws;
  auto alloc = [&](size_t bytes) -> void* {
    void* ptr = (void*)wsp;
    wsp += (bytes + 255) & ~(size_t)255;
    return ptr;
  };
  __bf16* xb   = (__bf16*)alloc((size_t)m * n * 2);
  __bf16* wb   = (__bf16*)alloc((size_t)p * n * 2);
  __bf16* Bxb0 = (__bf16*)alloc((size_t)m * ell * 2);
  __bf16* Bxb1 = (__bf16*)alloc((size_t)m * ell * 2);
  __bf16* Byb0 = (__bf16*)alloc((size_t)p * ell * 2);
  __bf16* Byb1 = (__bf16*)alloc((size_t)p * ell * 2);
  float*  Gx   = (float*)alloc(256 * 256 * 4);
  float*  Gy   = (float*)alloc(256 * 256 * 4);
  float*  Rf   = (float*)alloc(256 * 256 * 4);   // chol f32 scratch
  float*  Rif  = (float*)alloc(256 * 256 * 4);
  __bf16* Rxb  = (__bf16*)alloc(256 * 256 * 2);
  __bf16* Rxib = (__bf16*)alloc(256 * 256 * 2);
  __bf16* Ryb  = (__bf16*)alloc(256 * 256 * 2);
  __bf16* Ryib = (__bf16*)alloc(256 * 256 * 2);
  float*  Mc   = (float*)alloc(256 * 256 * 4);
  float*  Wt   = (float*)alloc(256 * 256 * 4);
  float*  Vt   = (float*)alloc(256 * 256 * 4);
  __bf16* WsTb = (__bf16*)alloc(128 * 256 * 2);
  __bf16* VsTb = (__bf16*)alloc(128 * 256 * 2);
  __bf16* TxTb = (__bf16*)alloc(128 * 256 * 2);
  __bf16* TyTb = (__bf16*)alloc(128 * 256 * 2);

  const dim3 b256(256);

  // one-time bf16 conversion of the streamed operands
  cvt_bf16<<<(m * n + 255) / 256, b256, 0, stream>>>(x, xb, m * n);
  cvt_bf16<<<(p * n + 255) / 256, b256, 0, stream>>>(w, wb, p * n);
  // FD fill phase: Bx = x[:, :ell], By = w[:, :ell]
  copy_block128_bf16<<<(m * ell + 255) / 256, b256, 0, stream>>>(x, n, Bxb0, m * ell);
  copy_block128_bf16<<<(p * ell + 255) / 256, b256, 0, stream>>>(w, n, Byb0, p * ell);

  __bf16* BxC = Bxb0; __bf16* BxN = Bxb1;
  __bf16* ByC = Byb0; __bf16* ByN = Byb1;

  for (int k = 1; k < nb; ++k) {
    const int colOff = k * ell;
    // Gram matrices of augmented sketches Cx=[BxC|x_blk], Cy=[ByC|w_blk]
    gram_wmma<<<dim3(16, 16), b256, 0, stream>>>(BxC, ell, xb, n, colOff, Gx, m);
    gram_wmma<<<dim3(16, 16), b256, 0, stream>>>(ByC, ell, wb, n, colOff, Gy, p);
    // Cholesky-QR factors + triangular inverses (f32 scratch, bf16 outputs)
    chol_inv<<<1, b256, 0, stream>>>(Gx, Rf, Rif, Rxb, Rxib);
    chol_inv<<<1, b256, 0, stream>>>(Gy, Rf, Rif, Ryb, Ryib);
    // core: Mc = Rx @ Ry^T   [256 x 256]
    gemm_abt_wmma<<<dim3(256 / 64, 256 / 128), b256, 0, stream>>>(
        Rxb, 256, Rxb, 256, 256, 0, Ryb, 256, Mc, nullptr, 256, 256, nullptr);
    // SVD + shrink + symmetric split
    jacobi_svd<<<1, b256, 0, stream>>>(Mc, Wt, Vt, WsTb, VsTb);
    // TxT = (Rxi @ Uscaled)^T = WsT @ Rxi^T   [128 x 256]; TyT likewise
    gemm_abt_wmma<<<dim3(256 / 64, 128 / 128), b256, 0, stream>>>(
        WsTb, 256, WsTb, 256, 256, 0, Rxib, 256, nullptr, TxTb, 256, 256, nullptr);
    gemm_abt_wmma<<<dim3(256 / 64, 128 / 128), b256, 0, stream>>>(
        VsTb, 256, VsTb, 256, 256, 0, Ryib, 256, nullptr, TyTb, 256, 256, nullptr);
    // sketch updates: BxN = Cx @ TxT^T   [m x 128]  (Q never materialized)
    gemm_abt_wmma<<<dim3(ell / 64, m / 128), b256, 0, stream>>>(
        BxC, ell, xb, n, ell, colOff, TxTb, 256, nullptr, BxN, ell, 256, nullptr);
    gemm_abt_wmma<<<dim3(ell / 64, p / 128), b256, 0, stream>>>(
        ByC, ell, wb, n, ell, colOff, TyTb, 256, nullptr, ByN, ell, 256, nullptr);
    __bf16* t;
    t = BxC; BxC = BxN; BxN = t;
    t = ByC; ByC = ByN; ByN = t;
  }

  // out = Bx @ By^T + bias   [m x p]
  gemm_abt_wmma<<<dim3(p / 64, m / 128), b256, 0, stream>>>(
      BxC, ell, BxC, ell, ell, 0, ByC, ell, out, nullptr, p, ell, bias);
}
